// HeteroGNN_79388175499540
// MI455X (gfx1250) — matrix operand order repaired and checked by reading, not compiled, not run on hardware
//
#include <hip/hip_runtime.h>
#include <hip/hip_bf16.h>
#include <stdint.h>

// ---------------------------------------------------------------------------
// HeteroGNN forward on gfx1250 (MI455X): bf16 WMMA GEMMs (4x register-blocked)
// + fp32 atomic scatter-mean aggregation + wave32 layernorm.
// Root transforms fused per destination type: sum_e(x @ Wr_e) = x @ sum_e(Wr_e).
// ---------------------------------------------------------------------------

#define NT 100000          // track nodes
#define NA 20000           // artist nodes
#define NG 1000            // genre nodes
#define HID 256            // hidden / out dim
#define KTRK 512           // track/artist input dim
#define KGEN 163           // genre input dim
#define KGENP 192          // genre input dim padded to multiple of 32
#define MT 4               // M-tiles (16 rows each) per wave in the GEMM

typedef __bf16 bf16_t;
typedef __attribute__((ext_vector_type(8)))  bf16_t v8bf;
typedef __attribute__((ext_vector_type(16))) bf16_t v16bf;
typedef __attribute__((ext_vector_type(8)))  float  v8f;

// ---------------------------- utility kernels ------------------------------

__global__ __launch_bounds__(256) void k_zero(float* __restrict__ p, long n) {
  long i = (long)blockIdx.x * blockDim.x + threadIdx.x;
  long st = (long)gridDim.x * blockDim.x;
  for (; i < n; i += st) p[i] = 0.0f;
}

// fp32 [rows x Kin] -> bf16 [rows x Kpad] with zero padding
__global__ __launch_bounds__(256) void k_cvt_bf16(const float* __restrict__ src,
                                                  bf16_t* __restrict__ dst,
                                                  long rows, int Kin, int Kpad) {
  long n = rows * (long)Kpad;
  long i = (long)blockIdx.x * blockDim.x + threadIdx.x;
  long st = (long)gridDim.x * blockDim.x;
  for (; i < n; i += st) {
    long r = i / Kpad;
    int k = (int)(i - r * (long)Kpad);
    float v = (k < Kin) ? src[r * (long)Kin + k] : 0.0f;
    dst[i] = (bf16_t)v;
  }
}

// fp32 W[K x N] -> bf16 Wt[N x Kpad] (transposed, zero padded along K)
__global__ __launch_bounds__(256) void k_cvt_wt(const float* __restrict__ W,
                                                bf16_t* __restrict__ Wt,
                                                int Kin, int N, int Kpad) {
  long n = (long)N * Kpad;
  long i = (long)blockIdx.x * blockDim.x + threadIdx.x;
  long st = (long)gridDim.x * blockDim.x;
  for (; i < n; i += st) {
    int col = (int)(i / Kpad);
    int k = (int)(i - (long)col * Kpad);
    float v = (k < Kin) ? W[(long)k * N + col] : 0.0f;
    Wt[i] = (bf16_t)v;
  }
}

// sum of up to 3 fp32 weight matrices [K x N] -> transposed bf16 [N x K]
// (root-transform fusion: HeteroConv 'sum' distributes over shared x_dst)
__global__ __launch_bounds__(256) void k_sum_wt(const float* __restrict__ W0,
                                                const float* __restrict__ W1,
                                                const float* __restrict__ W2,
                                                bf16_t* __restrict__ Wt, int N, int K) {
  long n = (long)N * K;
  long i = (long)blockIdx.x * blockDim.x + threadIdx.x;
  long st = (long)gridDim.x * blockDim.x;
  for (; i < n; i += st) {
    int col = (int)(i / K);
    int k = (int)(i - (long)col * K);
    long wi = (long)k * N + col;
    float v = W0[wi];
    if (W1) v += W1[wi];
    if (W2) v += W2[wi];
    Wt[i] = (bf16_t)v;
  }
}

// scatter-add of 256-wide source rows along edges; 64 threads per edge (4 floats each)
__global__ __launch_bounds__(256) void k_scatter(const float* __restrict__ x,
                                                 const int* __restrict__ ei,
                                                 int E, float* __restrict__ agg,
                                                 float* __restrict__ cnt) {
  long total = (long)E * 64;
  long i = (long)blockIdx.x * blockDim.x + threadIdx.x;
  long st = (long)gridDim.x * blockDim.x;
  for (; i < total; i += st) {
    int e = (int)(i >> 6);
    int f = (int)((i & 63) << 2);
    int s = ei[e];
    int d = ei[E + e];
    const float* xs = x + (long)s * HID + f;
    float* ag = agg + (long)d * HID + f;
    atomicAdd(ag + 0, xs[0]);
    atomicAdd(ag + 1, xs[1]);
    atomicAdd(ag + 2, xs[2]);
    atomicAdd(ag + 3, xs[3]);
    if (f == 0) atomicAdd(cnt + d, 1.0f);
  }
}

// mean = agg / max(cnt,1), emit bf16
__global__ __launch_bounds__(256) void k_mean_bf16(const float* __restrict__ agg,
                                                   const float* __restrict__ cnt,
                                                   bf16_t* __restrict__ out, long rows) {
  long n = rows * (long)HID;
  long i = (long)blockIdx.x * blockDim.x + threadIdx.x;
  long st = (long)gridDim.x * blockDim.x;
  for (; i < n; i += st) {
    long r = i >> 8;  // HID == 256
    float c = cnt[r];
    if (c < 1.0f) c = 1.0f;
    out[i] = (bf16_t)(agg[i] / c);
  }
}

// relu then layernorm, in place; one wave32 per 256-wide row
__global__ __launch_bounds__(256) void k_relu_ln(float* __restrict__ x,
                                                 const float* __restrict__ g,
                                                 const float* __restrict__ b, int rows) {
  int row = blockIdx.x * (blockDim.x >> 5) + (threadIdx.x >> 5);
  if (row >= rows) return;
  int lane = threadIdx.x & 31;
  float* xr = x + (long)row * HID;
  float v[8];
  float s = 0.0f;
#pragma unroll
  for (int j = 0; j < 8; ++j) {
    float t = xr[lane * 8 + j];
    t = t > 0.0f ? t : 0.0f;
    v[j] = t;
    s += t;
  }
#pragma unroll
  for (int o = 16; o > 0; o >>= 1) s += __shfl_xor(s, o, 32);
  float mu = s * (1.0f / HID);
  float q = 0.0f;
#pragma unroll
  for (int j = 0; j < 8; ++j) { float d = v[j] - mu; q += d * d; }
#pragma unroll
  for (int o = 16; o > 0; o >>= 1) q += __shfl_xor(q, o, 32);
  float inv = rsqrtf(q * (1.0f / HID) + 1e-5f);
#pragma unroll
  for (int j = 0; j < 8; ++j) {
    int c = lane * 8 + j;
    xr[c] = (v[j] - mu) * inv * g[c] + b[c];
  }
}

// ------------------------------ WMMA GEMM ----------------------------------
// C[M x N] (+)= A_bf16[M x Kp] @ Bt_bf16[N x Kp]^T  (+ bias) (relu)
// one wave computes a 64x16 C strip (MT=4 tiles of 16x16); the B fragment is
// loaded once per K-step and reused by 4 back-to-back v_wmma_f32_16x16x32_bf16.
__global__ __launch_bounds__(256) void k_gemm_bf16(const bf16_t* __restrict__ A,
                                                   const bf16_t* __restrict__ Bt,
                                                   float* __restrict__ C,
                                                   const float* __restrict__ bias,
                                                   int M, int N, int Kp,
                                                   int accum, int relu) {
  int wave = threadIdx.x >> 5;
  int lane = threadIdx.x & 31;
  int tilesN = N >> 4;
  int tilesM = (M + 15) >> 4;
  int tilesMg = (tilesM + MT - 1) / MT;
  int tile = blockIdx.x * (blockDim.x >> 5) + wave;
  if (tile >= tilesMg * tilesN) return;
  int tm = (tile / tilesN) * MT;   // first of MT M-tiles
  int tn = tile - (tile / tilesN) * tilesN;
  int i = lane & 15;
  int half = lane >> 4;

  const bf16_t* ap[MT];
#pragma unroll
  for (int t = 0; t < MT; ++t) {
    int r = (tm + t) * 16 + i;
    if (r >= M) r = M - 1;  // clamp; masked at store
    ap[t] = A + (long)r * Kp + half * 8;
  }
  const bf16_t* bp = Bt + (long)(tn * 16 + i) * Kp + half * 8;

  // warm the cache path for the strips we are about to stream (global_prefetch_b8)
#pragma unroll
  for (int t = 0; t < MT; ++t) {
    __builtin_prefetch((const void*)ap[t], 0, 3);
    __builtin_prefetch((const void*)(ap[t] + 256), 0, 3);
  }
  __builtin_prefetch((const void*)bp, 0, 3);

  v8f acc[MT];
#pragma unroll
  for (int t = 0; t < MT; ++t) acc[t] = (v8f){0.f, 0.f, 0.f, 0.f, 0.f, 0.f, 0.f, 0.f};

  for (int k = 0; k < Kp; k += 32) {
    // B frag (32x16 bf16): lane(col=i, half): K = 8*half+0..7 and 16+8*half+0..7
    v8bf b0 = *(const v8bf*)(bp + k);
    v8bf b1 = *(const v8bf*)(bp + k + 16);
    v16bf bv = __builtin_shufflevector(b0, b1, 0, 1, 2, 3, 4, 5, 6, 7,
                                       8, 9, 10, 11, 12, 13, 14, 15);
#pragma unroll
    for (int t = 0; t < MT; ++t) {
      // A frag (16x32 bf16): lane(row=i, half): K = 8*half+0..7 and 16+8*half+0..7
      v8bf a0 = *(const v8bf*)(ap[t] + k);
      v8bf a1 = *(const v8bf*)(ap[t] + k + 16);
      v16bf av = __builtin_shufflevector(a0, a1, 0, 1, 2, 3, 4, 5, 6, 7,
                                         8, 9, 10, 11, 12, 13, 14, 15);
      acc[t] = __builtin_amdgcn_wmma_f32_16x16x32_bf16(false, av, false, bv,
                                                       (short)0, acc[t], false, false);
    }
  }

  int col = tn * 16 + i;
  float badd = bias ? bias[col] : 0.0f;
#pragma unroll
  for (int t = 0; t < MT; ++t) {
#pragma unroll
    for (int r = 0; r < 8; ++r) {
      int row = (tm + t) * 16 + half * 8 + r;  // C/D layout: VGPR r -> M = r + 8*half
      if (row < M) {
        long idx = (long)row * N + col;
        float v = acc[t][r] + badd;
        if (accum) v += C[idx];
        if (relu) v = v > 0.0f ? v : 0.0f;
        C[idx] = v;
      }
    }
  }
}

// ------------------------------- host side ---------------------------------

static inline size_t al256(size_t x) { return (x + 255) & ~(size_t)255; }

extern "C" void kernel_launch(void* const* d_in, const int* in_sizes, int n_in,
                              void* d_out, int out_size, void* d_ws, size_t ws_size,
                              hipStream_t stream) {
  (void)in_sizes; (void)out_size; (void)ws_size;
  char* ws = (char*)d_ws;

  const float* xT = (const float*)d_in[0];
  const float* xA = (const float*)d_in[1];
  const float* xG = (const float*)d_in[2];
  const int* ei[7];
  for (int e = 0; e < 7; ++e) ei[e] = (const int*)d_in[3 + e];

  // EDGE_TYPES order (matches d_in[3..9]): made_by, rev_made_by, tagged,
  // rev_tagged, plays, rev_plays, co_genre.   type: 0=track 1=artist 2=genre
  static const int Ecnt[7] = {100000, 100000, 300000, 300000, 60000, 60000, 800000};
  static const int srcT[7] = {0, 1, 0, 2, 1, 2, 0};
  static const int dstT[7] = {1, 0, 2, 0, 2, 1, 0};
  // alphabetical (pytree) index of each edge name:
  // co_genre=0 made_by=1 plays=2 rev_made_by=3 rev_plays=4 rev_tagged=5 tagged=6
  static const int pidx[7] = {1, 3, 6, 5, 2, 4, 0};
  static const long typeOff[3] = {0, NT, NT + NA};
  static const int typeN[3] = {NT, NA, NG};

  // ---- param leaves in JAX pytree (sorted-key) order ----
  long leafSz[64];
  int li = 0;
  for (int s = 0; s < 7; ++s) { leafSz[li++] = 65536; leafSz[li++] = 65536; leafSz[li++] = 256; }
  for (int s = 0; s < 7; ++s) { leafSz[li++] = 65536; leafSz[li++] = 65536; leafSz[li++] = 256; }
  leafSz[li++] = 65536; leafSz[li++] = 256; leafSz[li++] = 65536; leafSz[li++] = 256;
  for (int s = 0; s < 6; ++s) leafSz[li++] = 256;
  for (int s = 0; s < 6; ++s) leafSz[li++] = 256;
  leafSz[li++] = 131072; leafSz[li++] = 256; leafSz[li++] = 41728; leafSz[li++] = 256;
  leafSz[li++] = 131072; leafSz[li++] = 256;
  long leafOff[64];
  long accp = 0;
  for (int t = 0; t < 64; ++t) { leafOff[t] = accp; accp += leafSz[t]; }
  const bool split = (n_in >= 10 + 64);
  const float* pflat = split ? nullptr : (const float*)d_in[10];
  auto P = [&](int leaf) -> const float* {
    return split ? (const float*)d_in[10 + leaf] : (pflat + leafOff[leaf]);
  };

  const float *c1Wl[7], *c1Wr[7], *c1bl[7], *c2Wl[7], *c2Wr[7], *c2bl[7];
  for (int e = 0; e < 7; ++e) {
    int s = pidx[e];
    c1Wl[e] = P(0 + 3 * s); c1Wr[e] = P(1 + 3 * s); c1bl[e] = P(2 + 3 * s);
    c2Wl[e] = P(21 + 3 * s); c2Wr[e] = P(22 + 3 * s); c2bl[e] = P(23 + 3 * s);
  }
  const float* headW1 = P(42); const float* headb1 = P(43);
  const float* headW2 = P(44); const float* headb2 = P(45);
  const float* n1b[3] = {P(50), P(46), P(48)};  // track, artist, genre
  const float* n1g[3] = {P(51), P(47), P(49)};
  const float* n2b[3] = {P(56), P(52), P(54)};
  const float* n2g[3] = {P(57), P(53), P(55)};
  const float* projW[3] = {P(62), P(58), P(60)};
  const float* projB[3] = {P(63), P(59), P(61)};

  // ---- workspace layout ----
  const long R = NT + NA + NG;
  size_t off = 0;
  size_t o_buf1 = off; off = al256(off + (size_t)R * HID * 4);
  size_t o_buf2 = off; off = al256(off + (size_t)R * HID * 4);
  size_t o_hbf  = off; off = al256(off + (size_t)R * HID * 2);
  // union region: {input bf16 copies (123.3MB)} then {agg fp32 + agg bf16 (153.6MB)}
  size_t o_D = off;   off = al256(off + (size_t)NT * HID * 4 + (size_t)NT * HID * 2);
  size_t o_cnt = off; off = al256(off + (size_t)NT * 4);
  size_t wo = off;
  auto walloc = [&](size_t elems) -> bf16_t* {
    size_t o = wo;
    wo = al256(wo + elems * 2);
    return (bf16_t*)(ws + o);
  };

  float* buf1 = (float*)(ws + o_buf1);
  float* buf2 = (float*)(ws + o_buf2);
  bf16_t* hbf = (bf16_t*)(ws + o_hbf);
  float* aggf = (float*)(ws + o_D);
  bf16_t* aggb = (bf16_t*)(ws + o_D + (size_t)NT * HID * 4);
  bf16_t* xTb = (bf16_t*)(ws + o_D);
  bf16_t* xAb = (bf16_t*)(ws + o_D + (size_t)NT * KTRK * 2);
  bf16_t* xGb = (bf16_t*)(ws + o_D + (size_t)(NT + NA) * KTRK * 2);
  float* cnt = (float*)(ws + o_cnt);

  bf16_t* projWt[3];
  projWt[0] = walloc((size_t)HID * KTRK);
  projWt[1] = walloc((size_t)HID * KTRK);
  projWt[2] = walloc((size_t)HID * KGENP);
  bf16_t *c1WlT[7], *c2WlT[7];
  for (int e = 0; e < 7; ++e) { c1WlT[e] = walloc(HID * HID); c2WlT[e] = walloc(HID * HID); }
  bf16_t* wrSumT[2][3];  // fused root transform per layer per dst type
  for (int L = 0; L < 2; ++L)
    for (int t = 0; t < 3; ++t) wrSumT[L][t] = walloc(HID * HID);
  bf16_t* hW1t = walloc(HID * HID);
  bf16_t* hW2t = walloc(HID * HID);

  // ---- launch helpers ----
  auto zero = [&](float* p, long n) {
    k_zero<<<(unsigned)((n + 255) / 256), 256, 0, stream>>>(p, n);
  };
  auto cvt = [&](const float* s, bf16_t* d, long rows, int Kin, int Kpad) {
    long n = rows * (long)Kpad;
    k_cvt_bf16<<<(unsigned)((n + 255) / 256), 256, 0, stream>>>(s, d, rows, Kin, Kpad);
  };
  auto cvtw = [&](const float* W, bf16_t* Wt, int Kin, int N, int Kpad) {
    long n = (long)N * Kpad;
    k_cvt_wt<<<(unsigned)((n + 255) / 256), 256, 0, stream>>>(W, Wt, Kin, N, Kpad);
  };
  auto sumw = [&](const float* W0, const float* W1, const float* W2, bf16_t* Wt) {
    long n = (long)HID * HID;
    k_sum_wt<<<(unsigned)((n + 255) / 256), 256, 0, stream>>>(W0, W1, W2, Wt, HID, HID);
  };
  auto gemm = [&](const bf16_t* A, const bf16_t* Bt, float* C, const float* bias,
                  int M, int N, int Kp, int accum, int relu) {
    long tilesM = (M + 15) / 16;
    long tiles = ((tilesM + MT - 1) / MT) * (N / 16);
    k_gemm_bf16<<<(unsigned)((tiles + 7) / 8), 256, 0, stream>>>(A, Bt, C, bias, M, N, Kp,
                                                                 accum, relu);
  };

  // ---- stage 0: weight + input bf16 conversions ----
  cvtw(projW[0], projWt[0], KTRK, HID, KTRK);
  cvtw(projW[1], projWt[1], KTRK, HID, KTRK);
  cvtw(projW[2], projWt[2], KGEN, HID, KGENP);
  for (int e = 0; e < 7; ++e) {
    cvtw(c1Wl[e], c1WlT[e], HID, HID, HID);
    cvtw(c2Wl[e], c2WlT[e], HID, HID, HID);
  }
  // fused Wr sums per dst type: track <- {rev_made_by, rev_tagged, co_genre},
  // artist <- {made_by, rev_plays}, genre <- {tagged, plays}
  sumw(c1Wr[1], c1Wr[3], c1Wr[6], wrSumT[0][0]);
  sumw(c1Wr[0], c1Wr[5], nullptr, wrSumT[0][1]);
  sumw(c1Wr[2], c1Wr[4], nullptr, wrSumT[0][2]);
  sumw(c2Wr[1], c2Wr[3], c2Wr[6], wrSumT[1][0]);
  sumw(c2Wr[0], c2Wr[5], nullptr, wrSumT[1][1]);
  sumw(c2Wr[2], c2Wr[4], nullptr, wrSumT[1][2]);
  cvtw(headW1, hW1t, HID, HID, HID);
  cvtw(headW2, hW2t, HID, HID, HID);
  cvt(xT, xTb, NT, KTRK, KTRK);
  cvt(xA, xAb, NA, KTRK, KTRK);
  cvt(xG, xGb, NG, KGEN, KGENP);

  // ---- stage 1: input projections + relu ----
  gemm(xTb, projWt[0], buf1 + typeOff[0] * HID, projB[0], NT, HID, KTRK, 0, 1);
  gemm(xAb, projWt[1], buf1 + typeOff[1] * HID, projB[1], NA, HID, KTRK, 0, 1);
  gemm(xGb, projWt[2], buf1 + typeOff[2] * HID, projB[2], NG, HID, KGENP, 0, 1);
  cvt(buf1, hbf, R, HID, HID);

  // ---- stages 2-3: two HeteroConv(SAGE) layers ----
  float* lin = buf1;
  float* lout = buf2;
  for (int L = 0; L < 2; ++L) {
    const float* const* bl = (L == 0) ? c1bl : c2bl;
    bf16_t* const* WlT = (L == 0) ? c1WlT : c2WlT;
    const float* const* nb = (L == 0) ? n1b : n2b;
    const float* const* ng = (L == 0) ? n1g : n2g;

    zero(lout, R * (long)HID);
    for (int e = 0; e < 7; ++e) {
      int sT = srcT[e], dT = dstT[e];
      int nd = typeN[dT];
      zero(aggf, (long)nd * HID);
      zero(cnt, nd);
      long tot = (long)Ecnt[e] * 64;
      k_scatter<<<(unsigned)((tot + 255) / 256), 256, 0, stream>>>(
          lin + typeOff[sT] * HID, ei[e], Ecnt[e], aggf, cnt);
      long nm = (long)nd * HID;
      k_mean_bf16<<<(unsigned)((nm + 255) / 256), 256, 0, stream>>>(aggf, cnt, aggb, nd);
      // out[dt] += mean @ Wl + bl
      gemm(aggb, WlT[e], lout + typeOff[dT] * HID, bl[e], nd, HID, HID, 1, 0);
    }
    // fused root transform: out[t] += x_t @ (sum_e Wr_e)
    for (int t = 0; t < 3; ++t) {
      gemm(hbf + typeOff[t] * HID, wrSumT[L][t], lout + typeOff[t] * HID, nullptr,
           typeN[t], HID, HID, 1, 0);
    }
    for (int t = 0; t < 3; ++t) {
      k_relu_ln<<<(typeN[t] + 7) / 8, 256, 0, stream>>>(lout + typeOff[t] * HID, ng[t], nb[t],
                                                        typeN[t]);
    }
    cvt(lout, hbf, R, HID, HID);
    float* tmp = lin; lin = lout; lout = tmp;
  }
  // after 2 layers: lin == final fp32 states, hbf == bf16 mirror (track at offset 0)

  // ---- stage 4: head on track embeddings ----
  float* t1 = buf2;  // reuse (track region)
  gemm(hbf, hW1t, t1, headb1, NT, HID, HID, 0, 1);
  cvt(t1, hbf, NT, HID, HID);
  gemm(hbf, hW2t, (float*)d_out, headb2, NT, HID, HID, 0, 0);
}
